// DeformConv3d_89481348645447
// MI455X (gfx1250) — matrix operand order repaired and checked by reading, not compiled, not run on hardware
//
#include <hip/hip_runtime.h>

typedef __attribute__((ext_vector_type(16))) __bf16 v16bf;
typedef __attribute__((ext_vector_type(8)))  float  v8f;

#define B_   4
#define T_   8
#define H_   56
#define W_   56
#define HW_  (H_*W_)            // 3136
#define N_   (B_*T_*HW_)        // 100352 spatial positions
#define C1   64
#define C2   128
#define OFFC 18
// padded staging geometry (zero halo of 1)
#define TP   (T_+2)             // 10
#define HP   (H_+2)             // 58
#define WP   (W_+2)             // 58
#define XPLANE  (HP*WP*C1)      // elems per (b,t) slab in padded x: 215296
#define XBATCH  (TP*XPLANE)     // elems per b
#define HPLANE  (HP*WP*C1)      // elems per bt slab in padded h

__device__ __forceinline__ unsigned short f2bf(float f){
  unsigned u = __builtin_bit_cast(unsigned, f);
  u += 0x7fffu + ((u >> 16) & 1u);        // round-to-nearest-even
  return (unsigned short)(u >> 16);
}
__device__ __forceinline__ float bflo(unsigned u){ return __builtin_bit_cast(float, u << 16); }
__device__ __forceinline__ float bfhi(unsigned u){ return __builtin_bit_cast(float, u & 0xffff0000u); }
__device__ __forceinline__ v8f zero8(){
  v8f z;
#pragma unroll
  for (int i = 0; i < 8; ++i) z[i] = 0.0f;
  return z;
}

// ---------------------------------------------------------------------------
// P0: x [B,C,T,H,W] fp32 -> padded xp [B,TP,HP,WP,C] bf16 interior.
// One block per (bt, h) row: reads coalesced per channel-row, writes one
// contiguous 7168B run. Halo stays zero from memset.
__global__ __launch_bounds__(256) void k_pack_x(const float* __restrict__ x,
                                                unsigned short* __restrict__ xp){
  __shared__ unsigned short tile[56 * 65];
  const int blk = blockIdx.x;                        // 32 bt * 56 rows
  const int bt = blk / 56;
  const int hh = blk % 56;
  const int t = bt & 7, b = bt >> 3;
  const float* src = x + ((size_t)b * C1 * T_ + t) * HW_ + hh * W_;
#pragma unroll
  for (int i = 0; i < 14; ++i){
    int e = threadIdx.x + 256 * i;                   // 3584 = 64c x 56p
    int c = e / 56, p = e - c * 56;                  // consecutive threads: consecutive p
    tile[p * 65 + c] = f2bf(src[(size_t)c * (T_ * HW_) + p]);
  }
  __syncthreads();
  unsigned short* dst = xp + (size_t)(b * TP + (t + 1)) * XPLANE
                           + (size_t)(hh + 1) * (WP * C1) + C1;  // w=1 start
#pragma unroll
  for (int i = 0; i < 14; ++i){
    int e = threadIdx.x + 256 * i;
    int p = e >> 6, c = e & 63;                      // consecutive threads: consecutive c
    dst[p * C1 + c] = tile[p * 65 + c];
  }
}

// P1: oc1_w [64,64,3,3,3] -> A-fragment stream: 54 ksteps x 4 mblocks x 512 bf16
__global__ __launch_bounds__(256) void k_pack_w1(const float* __restrict__ w,
                                                 unsigned short* __restrict__ wp){
  int tid = blockIdx.x * 256 + threadIdx.x;          // 110592
  if (tid >= 54 * 4 * 512) return;
  int e = tid & 511, blk = tid >> 9;
  int mb = blk & 3, s = blk >> 2;
  int lane = e >> 4, j = e & 15;
  int half = lane >> 4;
  int kk = half * 16 + j;
  int ti = s >> 1, ch = s & 1;
  int cin = ch * 32 + kk;
  int m = mb * 16 + (lane & 15);
  wp[tid] = f2bf(w[(m * C1 + cin) * 27 + ti]);
}

// P2: oc2_w [18,64,1,3,3] -> 18 ksteps x 2 mblocks x 512 (M padded to 32)
__global__ __launch_bounds__(256) void k_pack_w2(const float* __restrict__ w,
                                                 unsigned short* __restrict__ wp){
  int tid = blockIdx.x * 256 + threadIdx.x;          // 18432
  if (tid >= 18 * 2 * 512) return;
  int e = tid & 511, blk = tid >> 9;
  int mb = blk & 1, s = blk >> 1;
  int lane = e >> 4, j = e & 15;
  int half = lane >> 4;
  int kk = half * 16 + j;
  int ti = s >> 1, ch = s & 1;
  int cin = ch * 32 + kk;
  int m = mb * 16 + (lane & 15);
  float v = (m < OFFC) ? w[(m * C1 + cin) * 9 + ti] : 0.0f;
  wp[tid] = f2bf(v);
}

// P3: weight [128,64,3,3] -> 18 ksteps x 8 mblocks x 512
__global__ __launch_bounds__(256) void k_pack_w3(const float* __restrict__ w,
                                                 unsigned short* __restrict__ wp){
  int tid = blockIdx.x * 256 + threadIdx.x;          // 73728
  if (tid >= 18 * 8 * 512) return;
  int e = tid & 511, blk = tid >> 9;
  int mb = blk & 7, s = blk >> 3;
  int lane = e >> 4, j = e & 15;
  int half = lane >> 4;
  int kk = half * 16 + j;
  int ti = s >> 1, ch = s & 1;
  int cin = ch * 32 + kk;
  int m = mb * 16 + (lane & 15);
  wp[tid] = f2bf(w[(m * C1 + cin) * 9 + ti]);
}

// ---------------------------------------------------------------------------
// conv3d #1: implicit GEMM, per-WG tile 64x32. Padded input -> every B load is
// a plain b128 at a compile-time immediate offset; 54 back-to-back WMMAs.
// BN1 sum/sumsq fused into the epilogue (LDS atomics + 1 global atomic/chan).
__global__ __launch_bounds__(256) void k_conv1(const unsigned short* __restrict__ xp,
                                               const unsigned short* __restrict__ w1,
                                               const float* __restrict__ bias,
                                               float* __restrict__ h,
                                               float* __restrict__ gsum,
                                               float* __restrict__ gssq){
  __shared__ float ssum[C1], ssqr[C1];
  if (threadIdx.x < C1){ ssum[threadIdx.x] = 0.f; ssqr[threadIdx.x] = 0.f; }
  __syncthreads();

  const int lane = threadIdx.x & 31;
  const int wave = threadIdx.x >> 5;
  const int mb = wave & 3;            // M block (4 x 16 = 64)
  const int nb = wave >> 2;           // N block (2 x 16 = 32)
  const int n  = blockIdx.x * 32 + nb * 16 + (lane & 15);
  const int hw = n % HW_;
  const int bt = n / HW_;
  const int hh = hw / W_, ww = hw % W_;
  const int t = bt & 7, b = bt >> 3;
  const int cofs = (lane >> 4) << 4;
  const unsigned short* wbase = w1 + mb * 512 + lane * 16;
  // padded coords: data (t,h,w) lives at (t+1,h+1,w+1); tap (dz,dy,dx) reads
  // (t+dz, h+dy, w+dx) in padded space -> base at unshifted (t,hh,ww).
  const unsigned short* xbase = xp + (size_t)(b * TP + t) * XPLANE
                                   + (size_t)(hh * WP + ww) * C1 + cofs;

  v8f acc = zero8();
#pragma unroll
  for (int dz = 0; dz < 3; ++dz)
#pragma unroll
  for (int dy = 0; dy < 3; ++dy)
#pragma unroll
  for (int dx = 0; dx < 3; ++dx)
#pragma unroll
  for (int ch = 0; ch < 2; ++ch){
    const int s = ((dz * 3 + dy) * 3 + dx) * 2 + ch;           // constant
    v16bf A  = *(const v16bf*)(wbase + (size_t)s * (4 * 512));
    v16bf Bf = *(const v16bf*)(xbase + ((dz * HP + dy) * WP + dx) * C1 + ch * 32);
    acc = __builtin_amdgcn_wmma_f32_16x16x32_bf16(false, A, false, Bf,
                                                  (short)0, acc, false, false);
  }
  const int mbase = mb * 16 + ((lane >> 4) << 3);
  float* dst = h + (size_t)n * C1 + mbase;
#pragma unroll
  for (int v = 0; v < 8; ++v){
    float val = acc[v] + bias[mbase + v];
    dst[v] = val;
    atomicAdd(&ssum[mbase + v], val);
    atomicAdd(&ssqr[mbase + v], val * val);
  }
  __syncthreads();
  if (threadIdx.x < C1){
    atomicAdd(&gsum[threadIdx.x], ssum[threadIdx.x]);
    atomicAdd(&gssq[threadIdx.x], ssqr[threadIdx.x]);
  }
}

// ---------------------------------------------------------------------------
__global__ void k_finalize(const float* __restrict__ sum, const float* __restrict__ ssq,
                           const float* __restrict__ g, const float* __restrict__ b,
                           float* __restrict__ scale, float* __restrict__ shift,
                           int C, float invN){
  int c = blockIdx.x * blockDim.x + threadIdx.x;
  if (c < C){
    float m  = sum[c] * invN;
    float va = ssq[c] * invN - m * m;
    float rs = rsqrtf(va + 1e-5f);
    float sc = g[c] * rs;
    scale[c] = sc;
    shift[c] = b[c] - m * sc;
  }
}

// BN1 + ReLU fused into bf16 repack of h, written into padded hb interior
__global__ __launch_bounds__(256) void k_bnrelu_pack(const float* __restrict__ h,
                                                     const float* __restrict__ scale,
                                                     const float* __restrict__ shift,
                                                     unsigned short* __restrict__ hb){
  int idx = blockIdx.x * 256 + threadIdx.x;
  if (idx >= N_ * C1) return;
  int c = idx & 63;
  int n = idx >> 6;
  int hw = n % HW_;
  int bt = n / HW_;
  int hh = hw / W_, ww = hw % W_;
  float v = fmaxf(h[idx] * scale[c] + shift[c], 0.0f);
  hb[(size_t)bt * HPLANE + (size_t)((hh + 1) * WP + (ww + 1)) * C1 + c] = f2bf(v);
}

// ---------------------------------------------------------------------------
// offset conv: M=32 (18 real), K=576, per-WG tile 32x64; padded input -> no masks
__global__ __launch_bounds__(256) void k_conv2(const unsigned short* __restrict__ hb,
                                               const unsigned short* __restrict__ w2,
                                               const float* __restrict__ bias,
                                               float* __restrict__ off){
  const int lane = threadIdx.x & 31;
  const int wave = threadIdx.x >> 5;
  const int mb = wave & 1;
  const int nb = wave >> 1;           // 0..3
  const int n  = blockIdx.x * 64 + nb * 16 + (lane & 15);
  const int hw = n % HW_;
  const int bt = n / HW_;
  const int hh = hw / W_, ww = hw % W_;
  const int cofs = (lane >> 4) << 4;
  const unsigned short* wbase = w2 + mb * 512 + lane * 16;
  const unsigned short* hbase = hb + (size_t)bt * HPLANE
                                   + (size_t)(hh * WP + ww) * C1 + cofs;

  v8f acc = zero8();
#pragma unroll
  for (int dy = 0; dy < 3; ++dy)
#pragma unroll
  for (int dx = 0; dx < 3; ++dx)
#pragma unroll
  for (int ch = 0; ch < 2; ++ch){
    const int s = (dy * 3 + dx) * 2 + ch;            // constant
    v16bf A  = *(const v16bf*)(wbase + (size_t)s * (2 * 512));
    v16bf Bf = *(const v16bf*)(hbase + (dy * WP + dx) * C1 + ch * 32);
    acc = __builtin_amdgcn_wmma_f32_16x16x32_bf16(false, A, false, Bf,
                                                  (short)0, acc, false, false);
  }
  const int mbase = mb * 16 + ((lane >> 4) << 3);
#pragma unroll
  for (int v = 0; v < 8; ++v){
    int m = mbase + v;
    if (m < OFFC) off[(size_t)n * OFFC + m] = acc[v] + bias[m];
  }
}

// ---------------------------------------------------------------------------
// deformable conv: per-WG 16 positions x all 128 out channels; 9 taps x 2 chunks.
// Weights (144KB) async-DMA'd into LDS; bilinear B-frags (full tap = 64 cins)
// built in LDS per tap. Padded x slab: the zero ring absorbs boundary taps, so
// masks only guard slab bounds. BN2 sum/sumsq fused into epilogue.
__global__ __launch_bounds__(256) void k_deform(const unsigned short* __restrict__ xp,
                                                const unsigned short* __restrict__ w3,
                                                const float* __restrict__ off,
                                                const float* __restrict__ bias,
                                                float* __restrict__ outp,
                                                float* __restrict__ gsum,
                                                float* __restrict__ gssq){
  extern __shared__ unsigned short smem[];   // [1024 bfrag (2 chunks)][73728 weights]
  __shared__ float dsum[C2], dssq[C2];
  unsigned short* bS = smem;
  unsigned short* wS = smem + 1024;
  const int tid = threadIdx.x;
  if (tid < C2){ dsum[tid] = 0.f; dssq[tid] = 0.f; }

  // async copy all packed deform weights global -> LDS (ASYNCcnt path)
  {
    unsigned lbase = (unsigned)(unsigned long long)(void*)wS;
    for (int i = tid; i < 9216; i += 256){
      unsigned long long ga = (unsigned long long)(w3 + i * 8);   // 16B each
      unsigned la = lbase + i * 16;
      asm volatile("global_load_async_to_lds_b128 %0, %1, off" :: "v"(la), "v"(ga) : "memory");
    }
    asm volatile("s_wait_asynccnt 0" ::: "memory");
  }
  __syncthreads();

  const int lane = tid & 31;
  const int wave = tid >> 5;          // = M block 0..7
  const int n0 = blockIdx.x * 16;

  // builder role: position bp, cin-pair cp
  const int bp = tid & 15;
  const int cp = tid >> 4;            // 0..15
  const int bn = n0 + bp;
  const int bhw = bn % HW_;
  const int bbt = bn / HW_;
  const int bh = bhw / W_, bw = bhw % W_;
  const int b = bbt >> 3, t = bbt & 7;
  const unsigned short* xslab = xp + (size_t)(b * TP + (t + 1)) * XPLANE;
  const int kc   = cp * 2;                           // K index within a 32-chunk
  const int lgrp = (kc < 16) ? bp : (16 + bp);
  const int bidx = (lgrp * 16 + (kc & 15)) >> 1;     // u32 slot in chunk frag

  v8f acc = zero8();
#pragma unroll
  for (int ti = 0; ti < 9; ++ti){
    const int ky = ti / 3, kx = ti % 3;              // compile-time constants
    __syncthreads();
    {
      float oy = off[(size_t)bn * OFFC + ti * 2 + 0];
      float ox = off[(size_t)bn * OFFC + ti * 2 + 1];
      float y = oy + (float)(bh - 1 + ky);
      float x = ox + (float)(bw - 1 + kx);
      float fy = floorf(y), fx = floorf(x);
      int y0 = (int)fy, x0 = (int)fx;
      float ly = y - fy, lx = x - fx;
      float w00 = (1.f - ly) * (1.f - lx), w01 = (1.f - ly) * lx;
      float w10 = ly * (1.f - lx),         w11 = ly * lx;
      // padded row/col indices of the 4 corners; halo gives exact zeros,
      // predicates only guard slab bounds.
      const unsigned yp = (unsigned)(y0 + 1), xq = (unsigned)(x0 + 1);
      const unsigned yq = (unsigned)(y0 + 2), xr = (unsigned)(x0 + 2);
      bool v00 = yp < (unsigned)HP && xq < (unsigned)WP;
      bool v01 = yp < (unsigned)HP && xr < (unsigned)WP;
      bool v10 = yq < (unsigned)HP && xq < (unsigned)WP;
      bool v11 = yq < (unsigned)HP && xr < (unsigned)WP;
      int i00 = ((y0 + 1) * WP + (x0 + 1)) * C1;
      int i01 = i00 + C1;
      int i10 = i00 + WP * C1;
      int i11 = i10 + C1;
#pragma unroll
      for (int ch = 0; ch < 2; ++ch){
        const unsigned short* bp2 = xslab + ch * 32 + kc;
        unsigned p00 = v00 ? *(const unsigned*)(bp2 + i00) : 0u;
        unsigned p01 = v01 ? *(const unsigned*)(bp2 + i01) : 0u;
        unsigned p10 = v10 ? *(const unsigned*)(bp2 + i10) : 0u;
        unsigned p11 = v11 ? *(const unsigned*)(bp2 + i11) : 0u;
        float v0 = bflo(p00) * w00 + bflo(p01) * w01 + bflo(p10) * w10 + bflo(p11) * w11;
        float v1 = bfhi(p00) * w00 + bfhi(p01) * w01 + bfhi(p10) * w10 + bfhi(p11) * w11;
        unsigned pk = (unsigned)f2bf(v0) | ((unsigned)f2bf(v1) << 16);
        ((unsigned*)(bS + ch * 512))[bidx] = pk;
      }
    }
    __syncthreads();
#pragma unroll
    for (int ch = 0; ch < 2; ++ch){
      const int s = ti * 2 + ch;
      v16bf A  = *(const v16bf*)(wS + (size_t)(s * 8 + wave) * 512 + lane * 16);
      v16bf Bf = *(const v16bf*)(bS + ch * 512 + lane * 16);
      acc = __builtin_amdgcn_wmma_f32_16x16x32_bf16(false, A, false, Bf,
                                                    (short)0, acc, false, false);
    }
  }
  const int n = n0 + (lane & 15);
  const int mbase = wave * 16 + ((lane >> 4) << 3);
  float* dst = outp + (size_t)n * C2 + mbase;
#pragma unroll
  for (int v = 0; v < 8; ++v){
    float val = acc[v] + bias[mbase + v];
    dst[v] = val;
    atomicAdd(&dsum[mbase + v], val);
    atomicAdd(&dssq[mbase + v], val * val);
  }
  __syncthreads();
  if (tid < C2){
    atomicAdd(&gsum[tid], dsum[tid]);
    atomicAdd(&gssq[tid], dssq[tid]);
  }
}

// BN2 + ReLU + transpose channel-last -> [B,O,T,H,W], LDS-tiled for coalescing
__global__ __launch_bounds__(256) void k_out(const float* __restrict__ outp,
                                             const float* __restrict__ scale,
                                             const float* __restrict__ shift,
                                             float* __restrict__ out){
  __shared__ float tile[16 * 129];                   // 16 positions x 128 channels
  const int n0 = blockIdx.x * 16;                    // 3136 % 16 == 0
  const int bt = n0 / HW_;
  const int hw0 = n0 % HW_;
  const int t = bt & 7, b = bt >> 3;
#pragma unroll
  for (int i = 0; i < 8; ++i){
    int e = threadIdx.x + 256 * i;                   // 2048 elements
    int p = e >> 7, o = e & 127;                     // consecutive threads: consecutive o
    float v = outp[(size_t)(n0 + p) * C2 + o];
    tile[p * 129 + o] = fmaxf(v * scale[o] + shift[o], 0.0f);
  }
  __syncthreads();
  float* dst = out + ((size_t)b * C2 * T_ + t) * HW_ + hw0;
#pragma unroll
  for (int i = 0; i < 8; ++i){
    int e = threadIdx.x + 256 * i;
    int o = e >> 4, p = e & 15;                      // consecutive threads: consecutive p
    dst[(size_t)o * (T_ * HW_) + p] = tile[p * 129 + o];
  }
}

// ---------------------------------------------------------------------------
extern "C" void kernel_launch(void* const* d_in, const int* in_sizes, int n_in,
                              void* d_out, int out_size, void* d_ws, size_t ws_size,
                              hipStream_t stream){
  const float* x     = (const float*)d_in[0];
  const float* oc1_w = (const float*)d_in[1];
  const float* oc1_b = (const float*)d_in[2];
  const float* bn1_g = (const float*)d_in[3];
  const float* bn1_b = (const float*)d_in[4];
  const float* oc2_w = (const float*)d_in[5];
  const float* oc2_b = (const float*)d_in[6];
  const float* dfm_w = (const float*)d_in[7];
  const float* dfm_b = (const float*)d_in[8];
  const float* bn2_g = (const float*)d_in[9];
  const float* bn2_b = (const float*)d_in[10];

  char* ws = (char*)d_ws;
  size_t o = 0;
  const size_t XPsz = (size_t)B_ * TP * HP * WP * C1;        // 8,611,840 elems
  const size_t HBsz = (size_t)B_ * T_ * HP * WP * C1;        // 6,889,472 elems
  unsigned short* XP = (unsigned short*)(ws + o); o += XPsz * 2;                  // 17.2 MB
  float*          Hh = (float*)(ws + o);          o += (size_t)N_ * C1 * 4;       // 25.7 MB
  unsigned short* HB = (unsigned short*)(ws + o); o += HBsz * 2;                  // 13.8 MB
  float*          OF = (float*)(ws + o);          o += (size_t)N_ * OFFC * 4;     // 7.2 MB
  float*          OP = (float*)(ws + o);          o += (size_t)N_ * C2 * 4;       // 51.4 MB
  unsigned short* W1 = (unsigned short*)(ws + o); o += 110592 * 2;
  unsigned short* W2 = (unsigned short*)(ws + o); o += 18432 * 2;
  unsigned short* W3 = (unsigned short*)(ws + o); o += 73728 * 2;
  float*          ST = (float*)(ws + o);
  float* sum1 = ST;        float* ssq1 = ST + 64;
  float* sum2 = ST + 128;  float* ssq2 = ST + 256;
  float* sc1  = ST + 384;  float* sh1  = ST + 448;
  float* sc2  = ST + 512;  float* sh2  = ST + 640;

  hipMemsetAsync(ST, 0, 384 * sizeof(float), stream);   // zero sum/sumsq
  hipMemsetAsync(XP, 0, XPsz * 2, stream);              // zero halo
  hipMemsetAsync(HB, 0, HBsz * 2, stream);              // zero halo

  // pack
  k_pack_x <<<32 * 56, 256, 0, stream>>>(x, XP);
  k_pack_w1<<<(110592 + 255) / 256, 256, 0, stream>>>(oc1_w, W1);
  k_pack_w2<<<(18432 + 255) / 256, 256, 0, stream>>>(oc2_w, W2);
  k_pack_w3<<<(73728 + 255) / 256, 256, 0, stream>>>(dfm_w, W3);

  // conv1 (WMMA implicit GEMM, BN1 stats fused)
  k_conv1<<<N_ / 32, 256, 0, stream>>>(XP, W1, oc1_b, Hh, sum1, ssq1);
  k_finalize<<<1, 128, 0, stream>>>(sum1, ssq1, bn1_g, bn1_b, sc1, sh1, C1, 1.0f / (float)N_);
  k_bnrelu_pack<<<(N_ * C1 + 255) / 256, 256, 0, stream>>>(Hh, sc1, sh1, HB);

  // offset conv (WMMA)
  k_conv2<<<N_ / 64, 256, 0, stream>>>(HB, W2, oc2_b, OF);

  // deformable conv (async LDS weight DMA + bilinear B-frags + WMMA, BN2 stats fused)
  size_t shmem = (1024 + 73728) * sizeof(unsigned short);   // 149504 B < 320 KB
  k_deform<<<N_ / 16, 256, shmem, stream>>>(XP, W3, OF, dfm_b, OP, sum2, ssq2);

  // BN2 + ReLU + layout transpose to d_out
  k_finalize<<<1, 128, 0, stream>>>(sum2, ssq2, bn2_g, bn2_b, sc2, sh2, C2, 1.0f / (float)N_);
  k_out<<<N_ / 16, 256, 0, stream>>>(OP, sc2, sh2, (float*)d_out);
}
